// Conv_GINE_NN_9603546874322
// MI455X (gfx1250) — compile-verified, matrix operand order checked by reading, not run on hardware
//
#include <hip/hip_runtime.h>
#include <math.h>

// ---------------- model dims ----------------
constexpr int Hc     = 64;
constexpr int NKc    = 10;
constexpr int NLc    = 3;
constexpr int EDc    = 8;
constexpr int NNODEc = 3000;
constexpr int Tc     = 24;
constexpr int Bc     = 2;
constexpr int Mc     = 300;
constexpr int Ec     = 12000;
constexpr int OUTTc  = 12;
constexpr int GGc    = Bc * Tc;          // 48 graphs
constexpr long long ROWS_ALL = (long long)GGc * NNODEc;   // 144000
constexpr int TE_CHUNK = 1500;           // sequences per conv chunk

typedef __attribute__((ext_vector_type(16))) _Float16 v16h;
typedef __attribute__((ext_vector_type(8)))  _Float16 v8h;
typedef __attribute__((ext_vector_type(8)))  float    v8f;

#define ACT_NONE 0
#define ACT_RELU 1
#define ACT_SIG  2

// =====================================================================
// Weight packer: rearrange B(K,N) (or B^T given as (N,K)) into the exact
// 16x16x32-f16 WMMA B-fragment order, one v16h (32B) per (tile,chunk,lane).
// Zero-pads columns past Nc (used by the N=1 attention projection).
// =====================================================================
__global__ void pack_b(const float* __restrict__ Bm, _Float16* __restrict__ Bp,
                       int Nc, int Kd, int transB)
{
    long long idx = (long long)blockIdx.x * blockDim.x + threadIdx.x;
    int KC = Kd >> 5;
    long long total = (long long)((Nc + 15) >> 4) * KC * 512;
    if (idx >= total) return;
    int e  = (int)(idx & 15);
    int l  = (int)((idx >> 4) & 31);
    int kc = (int)((idx >> 9) % KC);
    int tn = (int)(idx / ((long long)KC * 512));
    int k  = (kc << 5) + ((e >> 3) << 4) + (((l >> 4) & 1) << 3) + (e & 7);
    int n  = (tn << 4) + (l & 15);
    float v = 0.f;
    if (n < Nc) v = transB ? Bm[(size_t)n * Kd + k] : Bm[(size_t)k * Nc + n];
    Bp[idx] = (_Float16)v;
}

// =====================================================================
// WMMA GEMM with packed-f16 B: C(M,N) = act(A(M,K) @ B + bias).
// REQUIRES M % 16 == 0 and K % 32 == 0 (true for every GEMM here), so the
// inner loop is branch-free: 4x global_load_b128 for A, 1x 32B load for B,
// cvt to f16, one v_wmma per 32-K chunk. One wave per 16x16 output tile.
// =====================================================================
__global__ __launch_bounds__(128)
void gemm_wmma(const float* __restrict__ A, const v16h* __restrict__ Bp,
               const float* __restrict__ bias, float* __restrict__ C,
               int Mr, int Nc, int Kd, int act)
{
    const int lane = threadIdx.x & 31;
    const int wave = threadIdx.x >> 5;
    const int ntn  = (Nc + 15) >> 4;
    const int tile = blockIdx.x * 4 + wave;
    const int tm   = tile / ntn;
    const int tn   = tile - tm * ntn;
    if (tm * 16 >= Mr) return;                 // uniform per wave
    const int m0 = tm << 4, n0 = tn << 4;
    const int half = (lane >> 4) & 1;
    const int ml   = lane & 15;
    const int KC   = Kd >> 5;

    const float* abase = A + (size_t)(m0 + ml) * Kd + (half << 3);
    const v16h*  bp    = Bp + (size_t)tn * KC * 32 + lane;

    v8f acc = {};
    for (int kc = 0; kc < KC; ++kc) {
        const float* ak = abase + (kc << 5);
        float4 a0 = *(const float4*)(ak);
        float4 a1 = *(const float4*)(ak + 4);
        float4 a2 = *(const float4*)(ak + 16);
        float4 a3 = *(const float4*)(ak + 20);
        v16h av;
        av[0]  = (_Float16)a0.x; av[1]  = (_Float16)a0.y;
        av[2]  = (_Float16)a0.z; av[3]  = (_Float16)a0.w;
        av[4]  = (_Float16)a1.x; av[5]  = (_Float16)a1.y;
        av[6]  = (_Float16)a1.z; av[7]  = (_Float16)a1.w;
        av[8]  = (_Float16)a2.x; av[9]  = (_Float16)a2.y;
        av[10] = (_Float16)a2.z; av[11] = (_Float16)a2.w;
        av[12] = (_Float16)a3.x; av[13] = (_Float16)a3.y;
        av[14] = (_Float16)a3.z; av[15] = (_Float16)a3.w;
        v16h bv = bp[(size_t)kc * 32];
        acc = __builtin_amdgcn_wmma_f32_16x16x32_f16(
                  false, av, false, bv, (short)0, acc, false, false);
    }
#pragma unroll
    for (int v = 0; v < 8; ++v) {
        int row = m0 + v + (half << 3);
        int col = n0 + ml;
        if (col < Nc) {
            float r = acc[v] + (bias ? bias[col] : 0.f);
            if (act == ACT_RELU)      r = fmaxf(r, 0.f);
            else if (act == ACT_SIG)  r = 1.f / (1.f + __expf(-r));
            C[(size_t)row * Nc + col] = r;
        }
    }
}

// =====================================================================
// Temporal-encoder front end: 10 parallel conv1d (k=1..19) + BN + ReLU
// producing the concatenated (rows, T, 640) feature matrix.
// =====================================================================
struct TEParams {
    const float* cw[NKc]; const float* cb[NKc];
    const float* bg[NKc]; const float* bb[NKc];
    const float* fc1w; const float* fc1b;
    const float* fc2w; const float* fc2b;
    const float* fc3w; const float* fc3b;
    const float* fbng; const float* fbnb;
};

__global__ void te_conv(const float* __restrict__ x, const int* __restrict__ mask,
                        int useMask, int seq0, int nseq, int perB,
                        TEParams tp, float* __restrict__ feat)
{
    long long idx = (long long)blockIdx.x * blockDim.x + threadIdx.x;
    long long total = (long long)nseq * Tc * (NKc * Hc);
    if (idx >= total) return;
    int col = (int)(idx % (NKc * Hc));
    int t   = (int)((idx / (NKc * Hc)) % Tc);
    int s   = (int)(idx / ((long long)(NKc * Hc) * Tc));
    int seq = seq0 + s;
    int b = seq / perB, i = seq - b * perB;
    int node = useMask ? mask[i] : i;
    const float* xs = x + ((size_t)b * NNODEc + node) * Tc;
    int k = col >> 6, c = col & 63;
    const float* w = tp.cw[k] + c * (2 * k + 1);
    float acc = tp.cb[k][c];
    for (int j = 0; j <= 2 * k; ++j) {
        int tt = t - k + j;
        if (tt >= 0 && tt < Tc) acc += w[j] * xs[tt];
    }
    acc = acc * (tp.bg[k][c] * rsqrtf(1.f + 1e-5f)) + tp.bb[k][c];
    feat[idx] = fmaxf(acc, 0.f);
}

// broadcast 'unmonitored' into xt layout (B*T, N, H)
__global__ void init_xt(const float* __restrict__ unmon, float* __restrict__ xt)
{
    long long idx = (long long)blockIdx.x * blockDim.x + threadIdx.x;
    long long total = (long long)Bc * Tc * NNODEc * Hc;
    if (idx >= total) return;
    int c = (int)(idx & 63);
    int n = (int)((idx >> 6) % NNODEc);
    long long q = idx / ((long long)64 * NNODEc);
    int t = (int)(q % Tc);
    xt[idx] = unmon[((size_t)n * Tc + t) * Hc + c];
}

// scatter monitored-node encodings (with final BN) into xt
__global__ void write_mon(const float* __restrict__ y, const int* __restrict__ mask,
                          int seq0, int nseq, const float* __restrict__ g,
                          const float* __restrict__ be, float* __restrict__ xt)
{
    long long idx = (long long)blockIdx.x * blockDim.x + threadIdx.x;
    long long total = (long long)nseq * Tc * Hc;
    if (idx >= total) return;
    int c = (int)(idx & 63);
    int t = (int)((idx >> 6) % Tc);
    int s = (int)(idx / ((long long)Tc * Hc));
    int seq = seq0 + s;
    int b = seq / Mc, mi = seq - b * Mc;
    int node = mask[mi];
    float v = y[idx] * (g[c] * rsqrtf(1.f + 1e-5f)) + be[c];
    xt[(((size_t)(b * Tc + t)) * NNODEc + node) * Hc + c] = v;
}

// write runoff encodings (with final BN) into xr layout (B*T, N, H)
__global__ void write_xr(const float* __restrict__ y, int seq0, int nseq,
                         const float* __restrict__ g, const float* __restrict__ be,
                         float* __restrict__ xr)
{
    long long idx = (long long)blockIdx.x * blockDim.x + threadIdx.x;
    long long total = (long long)nseq * Tc * Hc;
    if (idx >= total) return;
    int c = (int)(idx & 63);
    int t = (int)((idx >> 6) % Tc);
    int s = (int)(idx / ((long long)Tc * Hc));
    int seq = seq0 + s;
    int b = seq / NNODEc, n = seq - b * NNODEc;
    float v = y[idx] * (g[c] * rsqrtf(1.f + 1e-5f)) + be[c];
    xr[(((size_t)(b * Tc + t)) * NNODEc + n) * Hc + c] = v;
}

// per-layer GINE edge embeddings: eemb[l][e][c] = nan_to_num(ea) @ lin_w + lin_b
__global__ void edge_emb(const float* __restrict__ ea,
                         const float* lw0, const float* lb0,
                         const float* lw1, const float* lb1,
                         const float* lw2, const float* lb2,
                         float* __restrict__ eemb)
{
    long long idx = (long long)blockIdx.x * blockDim.x + threadIdx.x;
    long long total = (long long)NLc * Ec * Hc;
    if (idx >= total) return;
    int c = (int)(idx & 63);
    int e = (int)((idx >> 6) % Ec);
    int l = (int)(idx / ((long long)Ec * Hc));
    const float* lw = (l == 0) ? lw0 : (l == 1) ? lw1 : lw2;
    const float* lb = (l == 0) ? lb0 : (l == 1) ? lb1 : lb2;
    float acc = lb[c];
    for (int d = 0; d < EDc; ++d) {
        float v = ea[(size_t)e * EDc + d];
        if (isnan(v)) v = 0.f;
        else if (isinf(v)) v = (v > 0.f) ? 3.402823466e38f : -3.402823466e38f;
        acc += v * lw[d * Hc + c];
    }
    eemb[idx] = acc;
}

// GINE message scatter-add: agg[g][dst] += relu(h[g][src] + eemb)
__global__ void gine_scatter(const float* __restrict__ h, const float* __restrict__ ee,
                             const int* __restrict__ ei, float* __restrict__ agg)
{
    long long idx = (long long)blockIdx.x * blockDim.x + threadIdx.x;
    long long total = (long long)GGc * Ec * Hc;
    if (idx >= total) return;
    int c = (int)(idx & 63);
    int e = (int)((idx >> 6) % Ec);
    int g = (int)(idx / ((long long)Ec * Hc));
    int s = ei[e], d = ei[Ec + e];
    float m = h[((size_t)g * NNODEc + s) * Hc + c] + ee[(size_t)e * Hc + c];
    m = fmaxf(m, 0.f);
    unsafeAtomicAdd(&agg[((size_t)g * NNODEc + d) * Hc + c], m);
}

__global__ void ew_add(float* __restrict__ o, const float* __restrict__ a,
                       const float* __restrict__ b, long long n)
{
    long long i = (long long)blockIdx.x * blockDim.x + threadIdx.x;
    if (i < n) o[i] = a[i] + b[i];
}

// gated blend: h = a*h + (1-a)*hnew, a is per-row scalar
__global__ void blend(float* __restrict__ h, const float* __restrict__ hnew,
                      const float* __restrict__ a, long long n)
{
    long long i = (long long)blockIdx.x * blockDim.x + threadIdx.x;
    if (i < n) {
        float av = a[i >> 6];
        h[i] = av * h[i] + (1.f - av) * hnew[i];
    }
}

// (B*T, N, H) -> GRU order (N*B, T, H)
__global__ void to_gru(const float* __restrict__ h, float* __restrict__ gin)
{
    long long idx = (long long)blockIdx.x * blockDim.x + threadIdx.x;
    long long total = ROWS_ALL * Hc;
    if (idx >= total) return;
    int c = (int)(idx & 63);
    int t = (int)((idx >> 6) % Tc);
    long long q = idx / ((long long)64 * Tc);
    int b = (int)(q % Bc);
    int n = (int)(q / Bc);
    gin[idx] = h[(((size_t)(b * Tc + t)) * NNODEc + n) * Hc + c];
}

// GRU order (N*B, T, H) -> (B*T, N, H)
__global__ void from_gru(const float* __restrict__ y, float* __restrict__ o)
{
    long long idx = (long long)blockIdx.x * blockDim.x + threadIdx.x;
    long long total = ROWS_ALL * Hc;
    if (idx >= total) return;
    int c = (int)(idx & 63);
    int n = (int)((idx >> 6) % NNODEc);
    long long q = idx / ((long long)64 * NNODEc);
    int t = (int)(q % Tc);
    int b = (int)(q / Tc);
    o[idx] = y[(((size_t)(n * Bc + b)) * Tc + t) * Hc + c];
}

// concat [out2 | xr] -> (rows, 128)
__global__ void concat2(const float* __restrict__ a, const float* __restrict__ b,
                        float* __restrict__ o)
{
    long long idx = (long long)blockIdx.x * blockDim.x + threadIdx.x;
    long long total = ROWS_ALL * 2 * Hc;
    if (idx >= total) return;
    int k = (int)(idx & 127);
    long long r = idx >> 7;
    o[idx] = (k < Hc) ? a[r * Hc + k] : b[r * Hc + (k - Hc)];
}

// =====================================================================
// Fused GRU layer. Block = 128 threads (4 wave32), owns 16 rows.
// W_hh is packed into fragment-ordered f16 v16h's in LDS at entry; the
// recurrent state keeps an f16 shadow so A-fragments are 2x ds_load_b128.
// Each timestep: (16x64)@(64x192) = 24 WMMA ops + fused gates.
// =====================================================================
__global__ __launch_bounds__(128)
void gru_layer(const float* __restrict__ gi,   // (NR, T, 192)
               const float* __restrict__ whh,  // (192, 64)
               const float* __restrict__ bhh,  // (192)
               float* __restrict__ y,          // (NR, T, 64)
               int NR)
{
    __shared__ float    hs[16][Hc];          //  4 KB  f32 state (gate math)
    __shared__ _Float16 hs16[16][Hc];        //  2 KB  f16 shadow (WMMA A)
    __shared__ float    ghs[16][192];        // 12 KB  gh staging
    __shared__ v16h     whp[12 * 2 * 32];    // 24 KB  packed W_hh fragments
    __shared__ float    bh[192];
    const int tid  = threadIdx.x;
    const int lane = tid & 31, wave = tid >> 5;
    const int row0 = blockIdx.x * 16;
    const int half = (lane >> 4) & 1, ml = lane & 15;

    // pack W_hh into WMMA B-fragment order: whp[(nt*2+kc)*32+l][e] = Whh[j][k]
    {
        _Float16* wp = (_Float16*)whp;
        for (int i = tid; i < 12 * 2 * 512; i += 128) {
            int e  = i & 15;
            int l  = (i >> 4) & 31;
            int kc = (i >> 9) & 1;
            int nt = i >> 10;
            int k  = (kc << 5) + ((e >> 3) << 4) + (((l >> 4) & 1) << 3) + (e & 7);
            int j  = (nt << 4) + (l & 15);
            wp[i] = (_Float16)whh[j * Hc + k];
        }
    }
    for (int i = tid; i < 192; i += 128) bh[i] = bhh[i];
    for (int i = tid; i < 16 * Hc; i += 128) {
        hs[i >> 6][i & 63]   = 0.f;
        hs16[i >> 6][i & 63] = (_Float16)0.f;
    }
    __syncthreads();

    for (int t = 0; t < Tc; ++t) {
        // gh = h @ Whh^T + bhh
#pragma unroll
        for (int j = 0; j < 3; ++j) {
            const int nt = wave + 4 * j;
            v8f acc = {};
#pragma unroll
            for (int kc = 0; kc < 2; ++kc) {
                const _Float16* hp = &hs16[ml][(kc << 5) + (half << 3)];
                v8h lo = *(const v8h*)(hp);
                v8h hi = *(const v8h*)(hp + 16);
                v16h av = __builtin_shufflevector(lo, hi, 0, 1, 2, 3, 4, 5, 6, 7,
                                                  8, 9, 10, 11, 12, 13, 14, 15);
                v16h bv = whp[(nt * 2 + kc) * 32 + lane];
                acc = __builtin_amdgcn_wmma_f32_16x16x32_f16(
                          false, av, false, bv, (short)0, acc, false, false);
            }
#pragma unroll
            for (int v = 0; v < 8; ++v) {
                int rr = v + (half << 3);
                int cc = (nt << 4) + ml;
                ghs[rr][cc] = acc[v] + bh[cc];
            }
        }
        __syncthreads();
        // fused gates
        for (int i = tid; i < 16 * Hc; i += 128) {
            int rl = i >> 6, c = i & 63;
            int row = row0 + rl;
            if (row < NR) {
                const float* gp = gi + ((size_t)row * Tc + t) * 192;
                float ir = gp[c], iz = gp[64 + c], inn = gp[128 + c];
                float hr = ghs[rl][c], hz = ghs[rl][64 + c], hn = ghs[rl][128 + c];
                float rg = 1.f / (1.f + __expf(-(ir + hr)));
                float zg = 1.f / (1.f + __expf(-(iz + hz)));
                float nn = tanhf(inn + rg * hn);
                float hv = (1.f - zg) * nn + zg * hs[rl][c];
                hs[rl][c]   = hv;
                hs16[rl][c] = (_Float16)hv;
                y[((size_t)row * Tc + t) * Hc + c] = hv;
            }
        }
        __syncthreads();
    }
}

// final head: relu(h @ fin_w1 + b1) @ fin_w2 + b2, last 12 timesteps
__global__ void final_head(const float* __restrict__ yg,  // (N*B, T, 64)
                           const float* __restrict__ w1, const float* __restrict__ b1,
                           const float* __restrict__ w2, const float* __restrict__ b2,
                           float* __restrict__ out)       // (B*N, 12)
{
    long long idx = (long long)blockIdx.x * blockDim.x + threadIdx.x;
    long long total = (long long)Bc * NNODEc * OUTTc;
    if (idx >= total) return;
    int ot = (int)(idx % OUTTc);
    int n  = (int)((idx / OUTTc) % NNODEc);
    int b  = (int)(idx / ((long long)OUTTc * NNODEc));
    const float* hp = yg + (((size_t)(n * Bc + b)) * Tc + (Tc - OUTTc + ot)) * Hc;
    float o = b2[0];
    for (int j = 0; j < Hc / 2; ++j) {
        float z = b1[j];
        for (int k = 0; k < Hc; ++k) z += hp[k] * w1[k * (Hc / 2) + j];
        o += fmaxf(z, 0.f) * w2[j];
    }
    out[idx] = o;
}

// =====================================================================
// Host orchestration
// =====================================================================
static inline int cdiv(long long a, int b) { return (int)((a + b - 1) / b); }

extern "C" void kernel_launch(void* const* d_in, const int* in_sizes, int n_in,
                              void* d_out, int out_size, void* d_ws, size_t ws_size,
                              hipStream_t stream)
{
    // ---- input walk: setup_inputs() insertion order, params recursively ----
    int ip = 0;
    const float* norm_runoff = (const float*)d_in[ip++];
    const float* norm_h_x    = (const float*)d_in[ip++];
    const float* edge_attr   = (const float*)d_in[ip++];
    TEParams te1, te2;
    auto loadTE = [&](TEParams& tp) {
        for (int k = 0; k < NKc; ++k) tp.cw[k] = (const float*)d_in[ip++];
        for (int k = 0; k < NKc; ++k) tp.cb[k] = (const float*)d_in[ip++];
        for (int k = 0; k < NKc; ++k) tp.bg[k] = (const float*)d_in[ip++];
        for (int k = 0; k < NKc; ++k) tp.bb[k] = (const float*)d_in[ip++];
        tp.fc1w = (const float*)d_in[ip++]; tp.fc1b = (const float*)d_in[ip++];
        tp.fc2w = (const float*)d_in[ip++]; tp.fc2b = (const float*)d_in[ip++];
        tp.fc3w = (const float*)d_in[ip++]; tp.fc3b = (const float*)d_in[ip++];
        tp.fbng = (const float*)d_in[ip++]; tp.fbnb = (const float*)d_in[ip++];
    };
    loadTE(te1); loadTE(te2);
    const float* unmon = (const float*)d_in[ip++];
    struct GL { const float *lw, *lb, *w1, *b1, *w2, *b2; } gnn[NLc];
    for (int l = 0; l < NLc; ++l) {
        gnn[l].lw = (const float*)d_in[ip++]; gnn[l].lb = (const float*)d_in[ip++];
        gnn[l].w1 = (const float*)d_in[ip++]; gnn[l].b1 = (const float*)d_in[ip++];
        gnn[l].w2 = (const float*)d_in[ip++]; gnn[l].b2 = (const float*)d_in[ip++];
    }
    const float* aw1 = (const float*)d_in[ip++]; const float* ab1 = (const float*)d_in[ip++];
    const float* aw2 = (const float*)d_in[ip++]; const float* ab2 = (const float*)d_in[ip++];
    const float* wih0 = (const float*)d_in[ip++]; const float* whh0 = (const float*)d_in[ip++];
    const float* bih0 = (const float*)d_in[ip++]; const float* bhh0 = (const float*)d_in[ip++];
    const float* wih1 = (const float*)d_in[ip++]; const float* whh1 = (const float*)d_in[ip++];
    const float* bih1 = (const float*)d_in[ip++]; const float* bhh1 = (const float*)d_in[ip++];
    const float* trans_w = (const float*)d_in[ip++]; const float* trans_b = (const float*)d_in[ip++];
    const float* fin_w1 = (const float*)d_in[ip++]; const float* fin_b1 = (const float*)d_in[ip++];
    const float* fin_w2 = (const float*)d_in[ip++]; const float* fin_b2 = (const float*)d_in[ip++];
    const int* input_mask = (const int*)d_in[ip++];
    const int* edge_index = (const int*)d_in[ip++];
    (void)in_sizes; (void)n_in; (void)out_size; (void)ws_size;

    // ---- workspace layout (~620 MB of d_ws) ----
    float* w = (float*)d_ws;
    size_t off = 0;
    auto alloc = [&](size_t n) { float* p = w + off; off += n; return p; };
    float* feat = alloc((size_t)TE_CHUNK * Tc * NKc * Hc);  // 23.0M
    float* mA   = alloc((size_t)TE_CHUNK * Tc * Hc);
    float* mB   = alloc((size_t)TE_CHUNK * Tc * 32);
    float* mC   = alloc((size_t)TE_CHUNK * Tc * Hc);
    float* bufH = alloc((size_t)ROWS_ALL * Hc);             // xt / GNN state
    float* xr   = alloc((size_t)ROWS_ALL * Hc);
    float* agg  = alloc((size_t)ROWS_ALL * Hc);
    float* sum  = alloc((size_t)ROWS_ALL * Hc);
    float* zb   = alloc((size_t)ROWS_ALL * Hc);
    float* hnew = alloc((size_t)ROWS_ALL * Hc);
    float* t1b  = alloc((size_t)ROWS_ALL * Hc);
    float* avb  = alloc((size_t)ROWS_ALL);
    float* eemb = alloc((size_t)NLc * Ec * Hc);
    float* gin  = alloc((size_t)ROWS_ALL * Hc);
    float* g0   = alloc((size_t)ROWS_ALL * Hc);
    float* g1   = alloc((size_t)ROWS_ALL * Hc);
    float* giB  = alloc((size_t)ROWS_ALL * 3 * Hc);         // gi / concat (aliased)
    float* out2 = alloc((size_t)ROWS_ALL * Hc);
    float* cat  = giB;                                      // reuse (disjoint lifetime)

    // ---- pack every GEMM weight into WMMA fragment order (f16) ----
    auto pack = [&](const float* Bm, int Nc, int Kd, int transB) {
        size_t n = (size_t)((Nc + 15) / 16) * (Kd / 32) * 512;
        _Float16* p = (_Float16*)(w + off); off += (n + 1) / 2;
        pack_b<<<cdiv((long long)n, 256), 256, 0, stream>>>(Bm, p, Nc, Kd, transB);
        return (const _Float16*)p;
    };
    const _Float16 *p1fc1 = pack(te1.fc1w, Hc, NKc * Hc, 0),
                   *p1fc2 = pack(te1.fc2w, 32, Hc, 0),
                   *p1fc3 = pack(te1.fc3w, Hc, 32, 0);
    const _Float16 *p2fc1 = pack(te2.fc1w, Hc, NKc * Hc, 0),
                   *p2fc2 = pack(te2.fc2w, 32, Hc, 0),
                   *p2fc3 = pack(te2.fc3w, Hc, 32, 0);
    const _Float16 *pgw1[NLc], *pgw2[NLc];
    for (int l = 0; l < NLc; ++l) {
        pgw1[l] = pack(gnn[l].w1, Hc, Hc, 0);
        pgw2[l] = pack(gnn[l].w2, Hc, Hc, 0);
    }
    const _Float16 *paw1 = pack(aw1, Hc, Hc, 0),
                   *paw2 = pack(aw2, 1, Hc, 0),
                   *pwih0 = pack(wih0, 3 * Hc, Hc, 1),
                   *pwih1 = pack(wih1, 3 * Hc, Hc, 1),
                   *ptrans = pack(trans_w, Hc, 2 * Hc, 0);

    auto gemm = [&](const float* A, const _Float16* Bp, const float* bias, float* C,
                    long long Mr, int Nc, int Kd, int act) {
        long long tiles = ((Mr + 15) / 16) * ((Nc + 15) / 16);
        gemm_wmma<<<dim3((unsigned)((tiles + 3) / 4)), dim3(128), 0, stream>>>(
            A, (const v16h*)Bp, bias, C, (int)Mr, Nc, Kd, act);
    };

    // ---- temporal encoders (chunked conv -> WMMA MLP -> BN/scatter) ----
    {
        long long tot = (long long)Bc * Tc * NNODEc * Hc;
        init_xt<<<cdiv(tot, 256), 256, 0, stream>>>(unmon, bufH);
    }
    auto run_te = [&](const float* x, const int* mask, int useMask, int perB,
                      const TEParams& tp, const _Float16* pf1, const _Float16* pf2,
                      const _Float16* pf3, int kind /*0=mon->xt,1=xr*/, float* dst) {
        int totalSeq = useMask ? Bc * Mc : Bc * NNODEc;
        for (int s0 = 0; s0 < totalSeq; s0 += TE_CHUNK) {
            int ns = (totalSeq - s0 < TE_CHUNK) ? (totalSeq - s0) : TE_CHUNK;
            long long rows = (long long)ns * Tc;
            long long tot = rows * NKc * Hc;
            te_conv<<<cdiv(tot, 256), 256, 0, stream>>>(x, mask, useMask, s0, ns, perB, tp, feat);
            gemm(feat, pf1, tp.fc1b, mA, rows, Hc, NKc * Hc, ACT_RELU);
            gemm(mA,   pf2, tp.fc2b, mB, rows, 32, Hc, ACT_RELU);
            gemm(mB,   pf3, tp.fc3b, mC, rows, Hc, 32, ACT_NONE);
            long long wtot = rows * Hc;
            if (kind == 0)
                write_mon<<<cdiv(wtot, 256), 256, 0, stream>>>(mC, mask, s0, ns, tp.fbng, tp.fbnb, dst);
            else
                write_xr<<<cdiv(wtot, 256), 256, 0, stream>>>(mC, s0, ns, tp.fbng, tp.fbnb, dst);
        }
    };
    run_te(norm_h_x,    input_mask, 1, Mc,     te1, p1fc1, p1fc2, p1fc3, 0, bufH);
    run_te(norm_runoff, nullptr,    0, NNODEc, te2, p2fc1, p2fc2, p2fc3, 1, xr);

    {
        long long tot = (long long)NLc * Ec * Hc;
        edge_emb<<<cdiv(tot, 256), 256, 0, stream>>>(edge_attr,
            gnn[0].lw, gnn[0].lb, gnn[1].lw, gnn[1].lb, gnn[2].lw, gnn[2].lb, eemb);
    }

    const long long NE = ROWS_ALL * Hc;
    auto gnn_pass = [&](float* h) {
        for (int l = 0; l < NLc; ++l) {
            hipMemsetAsync(agg, 0, (size_t)NE * sizeof(float), stream);
            long long stot = (long long)GGc * Ec * Hc;
            gine_scatter<<<cdiv(stot, 256), 256, 0, stream>>>(h, eemb + (size_t)l * Ec * Hc,
                                                              edge_index, agg);
            ew_add<<<cdiv(NE, 256), 256, 0, stream>>>(sum, h, agg, NE);
            gemm(sum, pgw1[l], gnn[l].b1, zb,   ROWS_ALL, Hc, Hc, ACT_RELU);
            gemm(zb,  pgw2[l], gnn[l].b2, hnew, ROWS_ALL, Hc, Hc, ACT_NONE);
            gemm(h,   paw1,    ab1,       t1b,  ROWS_ALL, Hc, Hc, ACT_RELU);
            gemm(t1b, paw2,    ab2,       avb,  ROWS_ALL, 1,  Hc, ACT_SIG);
            blend<<<cdiv(NE, 256), 256, 0, stream>>>(h, hnew, avb, NE);
        }
    };
    auto gru_run = [&](const float* xin, const _Float16* pwih, const float* whh,
                       const float* bih, const float* bhh, float* yout) {
        gemm(xin, pwih, bih, giB, ROWS_ALL, 3 * Hc, Hc, ACT_NONE);
        gru_layer<<<dim3((NNODEc * Bc) / 16), dim3(128), 0, stream>>>(giB, whh, bhh, yout,
                                                                      NNODEc * Bc);
    };

    // ---- GNN pass 1 + GRU ----
    gnn_pass(bufH);
    to_gru<<<cdiv(NE, 256), 256, 0, stream>>>(bufH, gin);
    gru_run(gin, pwih0, whh0, bih0, bhh0, g0);
    gru_run(g0,  pwih1, whh1, bih1, bhh1, g1);
    from_gru<<<cdiv(NE, 256), 256, 0, stream>>>(g1, out2);

    // ---- transition: concat(out, xr) @ trans_w ----
    concat2<<<cdiv(NE * 2, 256), 256, 0, stream>>>(out2, xr, cat);
    gemm(cat, ptrans, trans_b, bufH, ROWS_ALL, Hc, 2 * Hc, ACT_NONE);

    // ---- GNN pass 2 + GRU ----
    gnn_pass(bufH);
    to_gru<<<cdiv(NE, 256), 256, 0, stream>>>(bufH, gin);
    gru_run(gin, pwih0, whh0, bih0, bhh0, g0);
    gru_run(g0,  pwih1, whh1, bih1, bhh1, g1);

    // ---- head ----
    {
        long long tot = (long long)Bc * NNODEc * OUTTc;
        final_head<<<cdiv(tot, 256), 256, 0, stream>>>(g1, fin_w1, fin_b1, fin_w2, fin_b2,
                                                       (float*)d_out);
    }
}
